// SelfAttention_81200651698592
// MI455X (gfx1250) — matrix operand order repaired and checked by reading, not compiled, not run on hardware
//
#include <hip/hip_runtime.h>

typedef _Float16 half_t;
typedef __attribute__((ext_vector_type(16))) _Float16 v16h;
typedef __attribute__((ext_vector_type(8)))  _Float16 v8h;
typedef __attribute__((ext_vector_type(8)))  float    v8f;

#define BATCH 2
#define SEQ   2048
#define EMB   1024
#define HEADS 16
#define DH    64
#define E3    3072
#define ROWS  (BATCH * SEQ)   // 4096

#define LDT 40      // padded LDS tile row stride (halves): 80B = 16B aligned, conflict-free
#define SCS 2056    // padded score row stride (floats)

// ---- WMMA fragment helpers (CDNA5 wave32; k order contiguous per lane, not the table order of CDNA5 ISA 7.12.2) ----

// 16-bit A matrix 16x32 (MxK). lane&15 = row M; lanes 0-15 hold K=0..15,
// lanes 16-31 hold K=16..31 (contiguous per lane).
__device__ __forceinline__ v16h frag_a_h(const half_t* p, int stride, int lane) {
  const int m  = lane & 15;
  const int kb = (lane & 16) ? 16 : 0;
  const half_t* r = p + (size_t)m * stride + kb;
  v8h lo = *(const v8h*)(r);
  v8h hi = *(const v8h*)(r + 8);
  v16h a;
#pragma unroll
  for (int i = 0; i < 8; ++i) { a[i] = lo[i]; a[i + 8] = hi[i]; }
  return a;
}

// Same A layout, but source is f32 (converted on the fly).
__device__ __forceinline__ v16h frag_a_f32(const float* p, int stride, int lane) {
  const int m  = lane & 15;
  const int kb = (lane & 16) ? 16 : 0;
  const float* r = p + (size_t)m * stride + kb;
  v16h a;
#pragma unroll
  for (int i = 0; i < 8; ++i) { a[i] = (half_t)r[i]; a[i + 8] = (half_t)r[i + 8]; }
  return a;
}

// 16-bit B matrix 32x16 (KxN), stored [n][k]: lane&15 = column N;
// lanes 0-15 hold K=0..15, lanes 16-31 hold K=16..31 (contiguous per lane).
__device__ __forceinline__ v16h frag_b_h(const half_t* p, int stride, int lane) {
  const int n  = lane & 15;
  const int kb = (lane & 16) ? 16 : 0;
  const half_t* r = p + (size_t)n * stride + kb;
  v8h lo = *(const v8h*)(r);
  v8h hi = *(const v8h*)(r + 8);
  v16h b;
#pragma unroll
  for (int i = 0; i < 8; ++i) { b[i] = lo[i]; b[i + 8] = hi[i]; }
  return b;
}

__device__ __forceinline__ v8f wmma_f16(v16h a, v16h b, v8f c) {
  return __builtin_amdgcn_wmma_f32_16x16x32_f16(false, a, false, b, (short)0, c,
                                                false, false);
}

// ---- Kernel 1: qkv = x @ w_in^T + b_in, scattered to head-major f16 Q/K/V ----
// Workgroup tile: 128 rows x 64 cols, 4 waves; each wave owns 32 rows (8 WMMAs/chunk).
__global__ void __launch_bounds__(128)
qkv_kernel(const float* __restrict__ x, const float* __restrict__ w_in,
           const float* __restrict__ b_in, half_t* __restrict__ Qb,
           half_t* __restrict__ Kb, half_t* __restrict__ Vb) {
  __shared__ alignas(16) half_t sA[128 * LDT];
  __shared__ alignas(16) half_t sB[64 * LDT];
  const int t = threadIdx.x;
  const int lane = t & 31, w = t >> 5;
  const int row0 = blockIdx.y * 128;
  const int col0 = blockIdx.x * 64;
  v8f acc[2][4] = {};

  const int ldrow = t >> 1;         // 0..63
  const int ldk   = (t & 1) * 16;   // 0 or 16
  const float* xsrc0 = x    + (size_t)(row0 + ldrow) * EMB + ldk;
  const float* xsrc1 = x    + (size_t)(row0 + 64 + ldrow) * EMB + ldk;
  const float* wsrc  = w_in + (size_t)(col0 + ldrow) * EMB + ldk;

  for (int k0 = 0; k0 < EMB; k0 += 32) {
    __builtin_prefetch(xsrc0 + k0 + 64, 0, 0);   // global_prefetch_b8
    __builtin_prefetch(xsrc1 + k0 + 64, 0, 0);
    __builtin_prefetch(wsrc  + k0 + 64, 0, 0);
    __syncthreads();
#pragma unroll
    for (int q = 0; q < 4; ++q) {
      float4 fa0 = *(const float4*)(xsrc0 + k0 + q * 4);
      float4 fa1 = *(const float4*)(xsrc1 + k0 + q * 4);
      float4 fb  = *(const float4*)(wsrc  + k0 + q * 4);
      half_t* da0 = &sA[ldrow * LDT + ldk + q * 4];
      half_t* da1 = &sA[(64 + ldrow) * LDT + ldk + q * 4];
      half_t* db  = &sB[ldrow * LDT + ldk + q * 4];
      da0[0] = (half_t)fa0.x; da0[1] = (half_t)fa0.y; da0[2] = (half_t)fa0.z; da0[3] = (half_t)fa0.w;
      da1[0] = (half_t)fa1.x; da1[1] = (half_t)fa1.y; da1[2] = (half_t)fa1.z; da1[3] = (half_t)fa1.w;
      db[0]  = (half_t)fb.x;  db[1]  = (half_t)fb.y;  db[2]  = (half_t)fb.z;  db[3]  = (half_t)fb.w;
    }
    __syncthreads();
    v16h af0 = frag_a_h(&sA[(w * 32) * LDT], LDT, lane);
    v16h af1 = frag_a_h(&sA[(w * 32 + 16) * LDT], LDT, lane);
#pragma unroll
    for (int j = 0; j < 4; ++j) {
      v16h bf = frag_b_h(&sB[(j * 16) * LDT], LDT, lane);
      acc[0][j] = wmma_f16(af0, bf, acc[0][j]);
      acc[1][j] = wmma_f16(af1, bf, acc[1][j]);
    }
  }

  const int nlane = lane & 15;
  const int hi8 = (lane & 16) ? 8 : 0;
#pragma unroll
  for (int j = 0; j < 4; ++j) {
    const int c = col0 + j * 16 + nlane;     // output column in [0, 3072)
    const float bias = b_in[c];
    const int sect = c >> 10;                // 0=Q 1=K 2=V
    const int e = c & 1023;
    const int hh = e >> 6, dd = e & 63;
    half_t* base = (sect == 0) ? Qb : ((sect == 1) ? Kb : Vb);
#pragma unroll
    for (int a = 0; a < 2; ++a) {
#pragma unroll
      for (int r = 0; r < 8; ++r) {
        const int row = row0 + w * 32 + a * 16 + r + hi8;   // in [0, 4096)
        const int bb = row >> 11, ns = row & 2047;
        base[(((size_t)bb * HEADS + hh) * SEQ + ns) * DH + dd] =
            (half_t)(acc[a][j][r] + bias);
      }
    }
  }
}

// ---- Kernel 2: attention for one (b, h, 16-query tile) ----
__global__ void __launch_bounds__(128)
attn_kernel(const half_t* __restrict__ Qb, const half_t* __restrict__ Kb,
            const half_t* __restrict__ Vb, half_t* __restrict__ Ob) {
  __shared__ alignas(16) float sc[16 * SCS];   // ~128.5 KB score tile (f32)
  __shared__ float redm[16][8];
  __shared__ float reds[16][8];
  __shared__ float sInv[16];

  const int t = threadIdx.x;
  const int lane = t & 31, w = t >> 5;
  const int q0 = blockIdx.x * 16;
  const int h = blockIdx.y, b = blockIdx.z;
  const size_t hoff = ((size_t)b * HEADS + h) * (size_t)SEQ * DH;
  const half_t* Qh = Qb + hoff;
  const half_t* Kh = Kb + hoff;
  const half_t* Vh = Vb + hoff;
  const int nlane = lane & 15;
  const int hi8 = (lane & 16) ? 8 : 0;

  // Q fragments for this 16-row tile: two 32-wide chunks of Dh=64
  const v16h aq0 = frag_a_h(Qh + (size_t)q0 * DH,      DH, lane);
  const v16h aq1 = frag_a_h(Qh + (size_t)q0 * DH + 32, DH, lane);
  const float scale = 0.125f;   // Dh^-0.5

  // Phase 1: scores = scale * Q @ K^T ; wave w owns keys [w*512, w*512+512)
  for (int g = 0; g < 32; ++g) {
    const int key0 = w * 512 + g * 16;
    __builtin_prefetch(Kh + (size_t)(key0 + 16) * DH, 0, 0);
    v16h bk0 = frag_b_h(Kh + (size_t)key0 * DH,      DH, lane);  // B(k,n)=K[key0+n][k]
    v16h bk1 = frag_b_h(Kh + (size_t)key0 * DH + 32, DH, lane);
    v8f s = {};
    s = wmma_f16(aq0, bk0, s);
    s = wmma_f16(aq1, bk1, s);
#pragma unroll
    for (int r = 0; r < 8; ++r)
      sc[(r + hi8) * SCS + key0 + nlane] = s[r] * scale;
  }
  __syncthreads();

  // Phase 2: softmax. 8 threads per row; exp written back in place (unnormalized).
  {
    const int row = t >> 3, sub = t & 7;
    float* srow = &sc[row * SCS];
    float pm = -3.0e38f;
    for (int j = sub; j < SEQ; j += 8) pm = fmaxf(pm, srow[j]);
    redm[row][sub] = pm;
    __syncthreads();
    float rmax = redm[row][0];
#pragma unroll
    for (int i = 1; i < 8; ++i) rmax = fmaxf(rmax, redm[row][i]);
    float ps = 0.f;
    for (int j = sub; j < SEQ; j += 8) {
      float ev = __expf(srow[j] - rmax);
      srow[j] = ev;
      ps += ev;
    }
    reds[row][sub] = ps;
    __syncthreads();
    float rs = 0.f;
#pragma unroll
    for (int i = 0; i < 8; ++i) rs += reds[row][i];
    if (sub == 0) sInv[row] = 1.0f / rs;
  }
  __syncthreads();

  // Phase 3: O = P @ V ; wave w owns d-columns [w*16, w*16+16)
  v8f o = {};
  const int kb16 = (lane & 16) ? 16 : 0;
  for (int c = 0; c < 64; ++c) {
    const int key0 = c * 32;
    v16h ap = frag_a_f32(&sc[key0], SCS, lane);   // P chunk 16x32, f32->f16
    v16h bv;                                      // B(k,n) = V[key0+k][w*16+n]
    const half_t* vp = Vh + (size_t)(key0 + kb16) * DH + w * 16 + nlane;
    __builtin_prefetch(vp + 32 * DH, 0, 0);
#pragma unroll
    for (int i = 0; i < 16; ++i) bv[i] = vp[(size_t)i * DH];
    o = wmma_f16(ap, bv, o);
  }
#pragma unroll
  for (int r = 0; r < 8; ++r) {
    const int m = r + hi8;
    const float val = o[r] * sInv[m];
    Ob[((size_t)b * SEQ + q0 + m) * EMB + h * DH + w * 16 + nlane] = (half_t)val;
  }
}

// ---- Kernel 3: out = O @ w_out^T + b_out (f32 output) ----
// Workgroup tile: 128 rows x 64 cols; A-fragments read directly from f16 O buffer.
__global__ void __launch_bounds__(128)
proj_kernel(const half_t* __restrict__ Ob, const float* __restrict__ w_out,
            const float* __restrict__ b_out, float* __restrict__ out) {
  __shared__ alignas(16) half_t sB[64 * LDT];
  const int t = threadIdx.x;
  const int lane = t & 31, w = t >> 5;
  const int row0 = blockIdx.y * 128;
  const int col0 = blockIdx.x * 64;
  v8f acc[2][4] = {};

  const int ldrow = t >> 1;
  const int ldk   = (t & 1) * 16;
  const float*  wsrc  = w_out + (size_t)(col0 + ldrow) * EMB + ldk;
  const half_t* abase = Ob + (size_t)(row0 + w * 32) * EMB;

  for (int k0 = 0; k0 < EMB; k0 += 32) {
    __builtin_prefetch(wsrc + k0 + 64, 0, 0);
    __syncthreads();
#pragma unroll
    for (int q = 0; q < 4; ++q) {
      float4 fb = *(const float4*)(wsrc + k0 + q * 4);
      half_t* db = &sB[ldrow * LDT + ldk + q * 4];
      db[0] = (half_t)fb.x; db[1] = (half_t)fb.y; db[2] = (half_t)fb.z; db[3] = (half_t)fb.w;
    }
    __syncthreads();
    v16h af0 = frag_a_h(abase + k0, EMB, lane);
    v16h af1 = frag_a_h(abase + (size_t)16 * EMB + k0, EMB, lane);
#pragma unroll
    for (int j = 0; j < 4; ++j) {
      v16h bf = frag_b_h(&sB[(j * 16) * LDT], LDT, lane);
      acc[0][j] = wmma_f16(af0, bf, acc[0][j]);
      acc[1][j] = wmma_f16(af1, bf, acc[1][j]);
    }
  }

  const int nlane = lane & 15;
  const int hi8 = (lane & 16) ? 8 : 0;
#pragma unroll
  for (int j = 0; j < 4; ++j) {
    const int c = col0 + j * 16 + nlane;
    const float bias = b_out[c];
#pragma unroll
    for (int a = 0; a < 2; ++a) {
#pragma unroll
      for (int r = 0; r < 8; ++r) {
        const int row = row0 + w * 32 + a * 16 + r + hi8;
        out[(size_t)row * EMB + c] = acc[a][j][r] + bias;
      }
    }
  }
}

extern "C" void kernel_launch(void* const* d_in, const int* in_sizes, int n_in,
                              void* d_out, int out_size, void* d_ws, size_t ws_size,
                              hipStream_t stream) {
  (void)in_sizes; (void)n_in; (void)out_size; (void)ws_size;
  const float* x     = (const float*)d_in[0];
  const float* w_in  = (const float*)d_in[1];
  const float* b_in  = (const float*)d_in[2];
  const float* w_out = (const float*)d_in[3];
  const float* b_out = (const float*)d_in[4];
  float* out = (float*)d_out;

  char* ws = (char*)d_ws;
  const size_t qkvBytes = (size_t)BATCH * HEADS * SEQ * DH * sizeof(half_t); // 8 MB
  half_t* Qb = (half_t*)(ws);
  half_t* Kb = (half_t*)(ws + qkvBytes);
  half_t* Vb = (half_t*)(ws + 2 * qkvBytes);
  half_t* Ob = (half_t*)(ws + 3 * qkvBytes);

  qkv_kernel <<<dim3(E3 / 64, ROWS / 128), 128, 0, stream>>>(x, w_in, b_in, Qb, Kb, Vb);
  attn_kernel<<<dim3(SEQ / 16, HEADS, BATCH), 128, 0, stream>>>(Qb, Kb, Vb, Ob);
  proj_kernel<<<dim3(EMB / 64, ROWS / 128), 128, 0, stream>>>(Ob, w_out, b_out, out);
}
